// GraphVAE_31172872634843
// MI455X (gfx1250) — compile-verified, matrix operand order checked by reading, not run on hardware
//
#include <hip/hip_runtime.h>
#include <math.h>

typedef __attribute__((ext_vector_type(16))) _Float16 v16h;
typedef __attribute__((ext_vector_type(8)))  _Float16 v8h;
typedef __attribute__((ext_vector_type(8)))  float    v8f;

constexpr int kBS = 16, kT = 50, kN = 100;
constexpr int kACT = 8, kHID = 64, kLAT = 16;
constexpr int kRows = kBS * kT * kN;          // 80000 tokens

// ---------------------------------------------------------------------------
// WMMA helpers (CDNA5 wave32, 16x16x32 f16 -> f32).
// 16-bit A layout (ISA 7.12.2): lane L holds row = L&15; its 16 halves are
// k in [hi*8, hi*8+8) and [16+hi*8, 16+hi*8+8), hi = L>>4: two contiguous
// 8-half runs => two b128 loads from a row-major tile. B mirrors A with
// lane = N col, so a TRANSPOSED (N-major) tile is gathered identically.
// C/D: VGPR r, lane L -> row = r + 8*(L>>4), col = L&15.
// ---------------------------------------------------------------------------
__device__ __forceinline__ v8f wmma16(v16h a, v16h b, v8f c) {
  return __builtin_amdgcn_wmma_f32_16x16x32_f16(false, a, false, b, (short)0, c,
                                                false, false);
}

// Row-major gather (A tiles, or transposed-B tiles); works for LDS or global.
__device__ __forceinline__ v16h load_frag_rm(const _Float16* tile, int ld, int lane) {
  const _Float16* p = tile + (lane & 15) * ld + (lane >> 4) * 8;
  v8h c0 = *(const v8h*)p;
  v8h c1 = *(const v8h*)(p + 16);
  v16h f;
#pragma unroll
  for (int i = 0; i < 8; ++i) { f[i] = c0[i]; f[8 + i] = c1[i]; }
  return f;
}

// A fragment straight from global f32 (row-major, lda % 8 == 0).
__device__ __forceinline__ v16h load_frag_a_g32(const float* A, int lda, int row0,
                                                int k0, int lane) {
  const float* p = A + (size_t)(row0 + (lane & 15)) * lda + k0 + (lane >> 4) * 8;
  float4 a0 = *(const float4*)p;
  float4 a1 = *(const float4*)(p + 4);
  float4 b0 = *(const float4*)(p + 16);
  float4 b1 = *(const float4*)(p + 20);
  v16h f;
  f[0] = (_Float16)a0.x;  f[1] = (_Float16)a0.y;  f[2] = (_Float16)a0.z;  f[3] = (_Float16)a0.w;
  f[4] = (_Float16)a1.x;  f[5] = (_Float16)a1.y;  f[6] = (_Float16)a1.z;  f[7] = (_Float16)a1.w;
  f[8] = (_Float16)b0.x;  f[9] = (_Float16)b0.y;  f[10] = (_Float16)b0.z; f[11] = (_Float16)b0.w;
  f[12] = (_Float16)b1.x; f[13] = (_Float16)b1.y; f[14] = (_Float16)b1.z; f[15] = (_Float16)b1.w;
  return f;
}

// B fragment from global row-major f32 [K][N] (scalar gather; once, for Whh).
__device__ __forceinline__ v16h load_frag_b_g32(const float* B, int ldb, int k0,
                                                int n0, int lane) {
  int col = n0 + (lane & 15), hi = lane >> 4;
  v16h f;
#pragma unroll
  for (int i = 0; i < 8; ++i) {
    int k = k0 + hi * 8 + i;
    f[i]     = (_Float16)B[(size_t)k * ldb + col];
    f[8 + i] = (_Float16)B[(size_t)(k + 16) * ldb + col];
  }
  return f;
}

__device__ __forceinline__ float sigmoidf(float x) { return 1.f / (1.f + __expf(-x)); }

// ---------------------------------------------------------------------------
// 0a) Pack concat input [o|a_prev|m|0] into f16 Xcat[80000][64] (one pass).
// ---------------------------------------------------------------------------
__global__ void k_xcat(const float* __restrict__ o, const float* __restrict__ ap,
                       const float* __restrict__ mk, _Float16* __restrict__ xc) {
  int i = blockIdx.x * 256 + threadIdx.x;        // (row, 8-col group)
  if (i >= kRows * 8) return;
  int row = i >> 3, grp = i & 7;
  v8h pk;
  const float* src;
  if (grp < 5)       src = o + (size_t)row * 40 + grp * 8;
  else if (grp == 5) src = ap + (size_t)row * 8;
  else if (grp == 6) src = mk + (size_t)row * 8;
  else               src = nullptr;
  if (src) {
    float4 u = *(const float4*)src, v = *(const float4*)(src + 4);
    pk[0] = (_Float16)u.x; pk[1] = (_Float16)u.y; pk[2] = (_Float16)u.z; pk[3] = (_Float16)u.w;
    pk[4] = (_Float16)v.x; pk[5] = (_Float16)v.y; pk[6] = (_Float16)v.z; pk[7] = (_Float16)v.w;
  } else {
#pragma unroll
    for (int q = 0; q < 8; ++q) pk[q] = (_Float16)0.f;
  }
  *(v8h*)&xc[(size_t)row * 64 + grp * 8] = pk;
}

// 0b) Adjacency bitmask: ab[r*4+d] covers cols d*32..d*32+31 of row r.
__global__ void k_adjbits(const int* __restrict__ adj, unsigned* __restrict__ abg) {
  int i = blockIdx.x * 256 + threadIdx.x;
  if (i >= 112 * 4) return;
  int r = i >> 2, d = i & 3;
  unsigned bits = 0u;
  if (r < kN)
    for (int j = 0; j < 32; ++j) {
      int jj = d * 32 + j;
      if (jj < kN && adj[r * kN + jj] != 0) bits |= (1u << j);
    }
  abg[i] = bits;
}

// ---------------------------------------------------------------------------
// 1) Embedding: emb = Xcat @ Wcat + (id_emb + bias_sum). Grid-stride M tiles.
//    id_emb+bias staged in LDS; %100 hoisted out of the epilogue.
// ---------------------------------------------------------------------------
__global__ __launch_bounds__(256, 1)
void k_emb(const _Float16* __restrict__ xc, const float* __restrict__ Wo,
           const float* __restrict__ Wp, const float* __restrict__ Wm,
           const float* __restrict__ bo, const float* __restrict__ bp,
           const float* __restrict__ bm, const float* __restrict__ ide,
           float* __restrict__ emb) {
  __shared__ __align__(16) _Float16 wT[64 * 64];  // wT[c][k] = Wcat[k][c]
  __shared__ float bsum[64];
  __shared__ __align__(16) float idel[kN * 64];   // id_emb + bias_sum
  int tid = threadIdx.x, lane = tid & 31, wave = tid >> 5;
  for (int i = tid; i < 64 * 64; i += 256) {
    int c = i & 63, k = i >> 6;
    float v = 0.f;
    if (k < 40)      v = Wo[k * 64 + c];
    else if (k < 48) v = Wp[(k - 40) * 64 + c];
    else if (k < 56) v = Wm[(k - 48) * 64 + c];
    wT[c * 64 + k] = (_Float16)v;
  }
  if (tid < 64) bsum[tid] = bo[tid] + bp[tid] + bm[tid];
  __syncthreads();
  for (int i = tid; i < kN * 64 / 4; i += 256) {
    float4 u = *(const float4*)&ide[i * 4];
    int c = (i * 4) & 63;
    float4 r = make_float4(u.x + bsum[c], u.y + bsum[c + 1],
                           u.z + bsum[c + 2], u.w + bsum[c + 3]);
    *(float4*)&idel[i * 4] = r;
  }
  __syncthreads();
  int crow = (lane >> 4) * 8, ccol = lane & 15;
  for (int mt = blockIdx.x * 8 + wave; mt < kRows / 16; mt += gridDim.x * 8) {
    int row0 = mt * 16;
    v8f acc[4] = {};
#pragma unroll
    for (int kk = 0; kk < 2; ++kk) {
      v16h a = load_frag_rm(xc + (size_t)row0 * 64 + kk * 32, 64, lane);
#pragma unroll
      for (int nt = 0; nt < 4; ++nt) {
        v16h b = load_frag_rm(&wT[(nt * 16) * 64 + kk * 32], 64, lane);
        acc[nt] = wmma16(a, b, acc[nt]);
      }
    }
    int rbase = row0 + crow;
    int node0 = rbase % kN;                       // one modulo per tile
#pragma unroll
    for (int nt = 0; nt < 4; ++nt)
#pragma unroll
      for (int r = 0; r < 8; ++r) {
        int node = node0 + r; if (node >= kN) node -= kN;
        int c = nt * 16 + ccol;
        emb[(size_t)(rbase + r) * 64 + c] = acc[nt][r] + idel[node * 64 + c];
      }
  }
}

// ---------------------------------------------------------------------------
// 2) LSTM input gates: G = emb @ Wih + bih + bhh  (M=80000, K=64, N=256).
// ---------------------------------------------------------------------------
__global__ __launch_bounds__(256, 1)
void k_gates(const float* __restrict__ emb, const float* __restrict__ Wih,
             const float* __restrict__ bih, const float* __restrict__ bhh,
             float* __restrict__ G) {
  __shared__ __align__(16) _Float16 wT[256 * 64];  // wT[n][k] = Wih[k][n]
  __shared__ float bsum[256];
  int tid = threadIdx.x, lane = tid & 31, wave = tid >> 5;
  for (int i = tid; i < 256 * 64; i += 256) {
    int c = i & 255, k = i >> 8;                   // coalesced global read over c
    wT[c * 64 + k] = (_Float16)Wih[k * 256 + c];
  }
  bsum[tid] = bih[tid] + bhh[tid];
  __syncthreads();
  int crow = (lane >> 4) * 8, ccol = lane & 15;
  for (int tile = blockIdx.x * 8 + wave; tile < (kRows / 16) * 4; tile += gridDim.x * 8) {
    int mt = tile >> 2, nc = tile & 3, row0 = mt * 16;
    v8f acc[4] = {};
#pragma unroll
    for (int kk = 0; kk < 2; ++kk) {
      v16h a = load_frag_a_g32(emb, 64, row0, kk * 32, lane);
#pragma unroll
      for (int nt = 0; nt < 4; ++nt) {
        v16h b = load_frag_rm(&wT[(nc * 64 + nt * 16) * 64 + kk * 32], 64, lane);
        acc[nt] = wmma16(a, b, acc[nt]);
      }
    }
#pragma unroll
    for (int nt = 0; nt < 4; ++nt)
#pragma unroll
      for (int r = 0; r < 8; ++r) {
        int rg = row0 + crow + r, c = nc * 64 + nt * 16 + ccol;
        G[(size_t)rg * 256 + c] = acc[nt][r] + bsum[c];
      }
  }
}

// ---------------------------------------------------------------------------
// 3) LSTM recurrence. 128 thr = 4 waves = 16 sequences; wave w owns gate
//    chunk [64w,64w+64). Whh B-fragments pinned in VGPRs for all 50 steps.
// ---------------------------------------------------------------------------
__global__ __launch_bounds__(128, 1)
void k_lstm(const float* __restrict__ G, const float* __restrict__ Whh,
            float* __restrict__ ys) {
  __shared__ __align__(16) _Float16 hl[16 * 64];
  __shared__ __align__(16) float gl[16 * 256];
  int tid = threadIdx.x, lane = tid & 31, wave = tid >> 5;
  int s0 = blockIdx.x * 16;
  v16h bf[8];
#pragma unroll
  for (int kk = 0; kk < 2; ++kk)
#pragma unroll
    for (int nt = 0; nt < 4; ++nt)
      bf[kk * 4 + nt] = load_frag_b_g32(Whh, 256, kk * 32, wave * 64 + nt * 16, lane);
  for (int i = tid; i < 16 * 64; i += 128) hl[i] = (_Float16)0.f;
  float c8[8];
#pragma unroll
  for (int j = 0; j < 8; ++j) c8[j] = 0.f;
  __syncthreads();
  int crow = (lane >> 4) * 8, ccol = lane & 15;
  int e0 = tid * 8, erow = e0 >> 6, ecol = e0 & 63;
  int eseq = s0 + erow, eb = eseq / kN, en = eseq % kN;
  size_t gbase = (size_t)((eb * kT) * kN + en) * 256 + ecol;   // + t*25600
  size_t ybase = (size_t)((eb * kT) * kN + en) * 64 + ecol;    // + t*6400
  for (int t = 0; t < kT; ++t) {
    v8f acc[4] = {};
#pragma unroll
    for (int kk = 0; kk < 2; ++kk) {
      v16h a = load_frag_rm(&hl[kk * 32], 64, lane);
#pragma unroll
      for (int nt = 0; nt < 4; ++nt) acc[nt] = wmma16(a, bf[kk * 4 + nt], acc[nt]);
    }
#pragma unroll
    for (int nt = 0; nt < 4; ++nt)
#pragma unroll
      for (int r = 0; r < 8; ++r)
        gl[(crow + r) * 256 + wave * 64 + nt * 16 + ccol] = acc[nt][r];
    __syncthreads();
    {
      const float* gr = &gl[erow * 256 + ecol];
      const float* Gp = G + gbase + (size_t)t * (kN * 256);
      float4 i0 = *(const float4*)(gr),       i1 = *(const float4*)(gr + 4);
      float4 f0 = *(const float4*)(gr + 64),  f1 = *(const float4*)(gr + 68);
      float4 g0 = *(const float4*)(gr + 128), g1 = *(const float4*)(gr + 132);
      float4 o0 = *(const float4*)(gr + 192), o1 = *(const float4*)(gr + 196);
      float4 I0 = *(const float4*)(Gp),       I1 = *(const float4*)(Gp + 4);
      float4 F0 = *(const float4*)(Gp + 64),  F1 = *(const float4*)(Gp + 68);
      float4 G0 = *(const float4*)(Gp + 128), G1 = *(const float4*)(Gp + 132);
      float4 O0 = *(const float4*)(Gp + 192), O1 = *(const float4*)(Gp + 196);
      float gi[8] = {i0.x + I0.x, i0.y + I0.y, i0.z + I0.z, i0.w + I0.w,
                     i1.x + I1.x, i1.y + I1.y, i1.z + I1.z, i1.w + I1.w};
      float gf[8] = {f0.x + F0.x, f0.y + F0.y, f0.z + F0.z, f0.w + F0.w,
                     f1.x + F1.x, f1.y + F1.y, f1.z + F1.z, f1.w + F1.w};
      float gg[8] = {g0.x + G0.x, g0.y + G0.y, g0.z + G0.z, g0.w + G0.w,
                     g1.x + G1.x, g1.y + G1.y, g1.z + G1.z, g1.w + G1.w};
      float go[8] = {o0.x + O0.x, o0.y + O0.y, o0.z + O0.z, o0.w + O0.w,
                     o1.x + O1.x, o1.y + O1.y, o1.z + O1.z, o1.w + O1.w};
      v8h hpack;
      float hv[8];
#pragma unroll
      for (int j = 0; j < 8; ++j) {
        float cc = sigmoidf(gf[j]) * c8[j] + sigmoidf(gi[j]) * tanhf(gg[j]);
        float hh = sigmoidf(go[j]) * tanhf(cc);
        c8[j] = cc;
        hv[j] = hh;
        hpack[j] = (_Float16)hh;
      }
      *(v8h*)&hl[erow * 64 + ecol] = hpack;
      float* yp = &ys[ybase + (size_t)t * (kN * 64)];
      *(float4*)yp       = make_float4(hv[0], hv[1], hv[2], hv[3]);
      *(float4*)(yp + 4) = make_float4(hv[4], hv[5], hv[6], hv[7]);
    }
    __syncthreads();
  }
}

// ---------------------------------------------------------------------------
// 4) GAT layer, one workgroup (4 waves) per graph. n=100 padded to 112 rows /
//    K=128. h kept TRANSPOSED in LDS (hT[n][node]). CONCAT => +bias, ELU.
// ---------------------------------------------------------------------------
template <int FIN, int FINP, int H, int FOUT, int FOUTP, bool CONCAT>
__global__ __launch_bounds__(128, 1)
void k_gat(const float* __restrict__ x, const float* __restrict__ w,
           const float* __restrict__ as, const float* __restrict__ ad,
           const float* __restrict__ bias, const unsigned* __restrict__ abg,
           float* __restrict__ out) {
  constexpr int OW = CONCAT ? H * FOUT : FOUT;
  __shared__ __align__(16) _Float16 xl[112 * FINP];
  __shared__ __align__(16) _Float16 hT[FOUTP * 128];   // hT[n][node]
  __shared__ __align__(16) _Float16 wT[FOUTP * FINP];  // wT[n][k]
  __shared__ __align__(16) _Float16 att[4][16 * 128];
  __shared__ __align__(16) float es[128], ed[128];
  __shared__ float asl[FOUT], adl[FOUT], bl[OW];
  __shared__ unsigned ab[112 * 4];
  int tid = threadIdx.x, lane = tid & 31, wave = tid >> 5;
  int g = blockIdx.x;
  // stage x as f16 (8 cols per iteration; FIN is a multiple of 8)
  for (int i = tid; i < 112 * (FINP / 8); i += 128) {
    int r = i / (FINP / 8), c0 = (i % (FINP / 8)) * 8;
    v8h pk;
    if (r < kN && c0 < FIN) {
      const float* src = &x[((size_t)g * kN + r) * FIN + c0];
      float4 u = *(const float4*)src, v = *(const float4*)(src + 4);
      pk[0] = (_Float16)u.x; pk[1] = (_Float16)u.y; pk[2] = (_Float16)u.z; pk[3] = (_Float16)u.w;
      pk[4] = (_Float16)v.x; pk[5] = (_Float16)v.y; pk[6] = (_Float16)v.z; pk[7] = (_Float16)v.w;
    } else {
#pragma unroll
      for (int q = 0; q < 8; ++q) pk[q] = (_Float16)0.f;
    }
    *(v8h*)&xl[r * FINP + c0] = pk;
  }
  for (int i = tid; i < 112 * 4; i += 128) ab[i] = abg[i];
  if (tid < OW) bl[tid] = bias[tid];
  // zero pad nodes 112..127 of hT (never written by the 7 M tiles)
  for (int i = tid; i < FOUTP * 16; i += 128) hT[(i >> 4) * 128 + 112 + (i & 15)] = (_Float16)0.f;
  __syncthreads();

  for (int hd = 0; hd < H; ++hd) {
    for (int i = tid; i < FOUTP * FINP; i += 128) {
      int n = i / FINP, k = i % FINP;
      float v = (k < FIN && n < FOUT) ? w[((size_t)hd * FIN + k) * FOUT + n] : 0.f;
      wT[n * FINP + k] = (_Float16)v;
    }
    if (tid < FOUT) { asl[tid] = as[hd * FOUT + tid]; adl[tid] = ad[hd * FOUT + tid]; }
    __syncthreads();
    // h = x @ w[hd]  -> hT[n][node]
    for (int it = 0; it < 2; ++it) {
      int mt = wave + it * 4;
      if (mt < 7) {
        v8f acc[FOUTP / 16] = {};
#pragma unroll
        for (int kk = 0; kk < FINP / 32; ++kk) {
          v16h a = load_frag_rm(&xl[(mt * 16) * FINP + kk * 32], FINP, lane);
#pragma unroll
          for (int nt = 0; nt < FOUTP / 16; ++nt) {
            v16h b = load_frag_rm(&wT[(nt * 16) * FINP + kk * 32], FINP, lane);
            acc[nt] = wmma16(a, b, acc[nt]);
          }
        }
        int crow = (lane >> 4) * 8, ccol = lane & 15;
#pragma unroll
        for (int nt = 0; nt < FOUTP / 16; ++nt) {
          v8h pk;
#pragma unroll
          for (int r = 0; r < 8; ++r) pk[r] = (_Float16)acc[nt][r];
          *(v8h*)&hT[(nt * 16 + ccol) * 128 + mt * 16 + crow] = pk;
        }
      }
    }
    __syncthreads();
    // attention logits per node
    for (int nd = tid; nd < 128; nd += 128) {
      float ss = 0.f, dd = 0.f;
      if (nd < kN)
#pragma unroll
        for (int o2 = 0; o2 < FOUT; ++o2) {
          float hv = (float)hT[o2 * 128 + nd];
          ss += hv * asl[o2];
          dd += hv * adl[o2];
        }
      es[nd] = ss; ed[nd] = dd;
    }
    __syncthreads();
    // masked softmax rows (2-pass, shift-free: logits are O(1)) -> f16 A tiles
    for (int it = 0; it < 2; ++it) {
      int mt = wave + it * 4;
      if (mt < 7 && lane < 16) {
        int ri = mt * 16 + lane;
        _Float16* ar = &att[wave][lane * 128];
        if (ri < kN) {
          unsigned abr[4] = {ab[ri * 4], ab[ri * 4 + 1], ab[ri * 4 + 2], ab[ri * 4 + 3]};
          float rs = es[ri], sum = 0.f;
#pragma unroll 1
          for (int j0 = 0; j0 < 128; j0 += 8) {        // pass 1: sum of exp
            float4 e0 = *(const float4*)&ed[j0], e1 = *(const float4*)&ed[j0 + 4];
            float ev[8] = {e0.x, e0.y, e0.z, e0.w, e1.x, e1.y, e1.z, e1.w};
#pragma unroll
            for (int q = 0; q < 8; ++q) {
              int j = j0 + q;
              float e = rs + ev[q];
              e = e > 0.f ? e : 0.2f * e;
              unsigned bit = (abr[j >> 5] >> (j & 31)) & 1u;
              sum += bit ? __expf(e) : 0.f;
            }
          }
          float inv = 1.f / sum;
#pragma unroll 1
          for (int j0 = 0; j0 < 128; j0 += 8) {        // pass 2: write normalized
            float4 e0 = *(const float4*)&ed[j0], e1 = *(const float4*)&ed[j0 + 4];
            float ev[8] = {e0.x, e0.y, e0.z, e0.w, e1.x, e1.y, e1.z, e1.w};
            v8h pk;
#pragma unroll
            for (int q = 0; q < 8; ++q) {
              int j = j0 + q;
              float e = rs + ev[q];
              e = e > 0.f ? e : 0.2f * e;
              unsigned bit = (abr[j >> 5] >> (j & 31)) & 1u;
              pk[q] = (_Float16)(bit ? __expf(e) * inv : 0.f);
            }
            *(v8h*)&ar[j0] = pk;
          }
        } else {
          v8h pk;
#pragma unroll
          for (int q = 0; q < 8; ++q) pk[q] = (_Float16)0.f;
#pragma unroll 1
          for (int j0 = 0; j0 < 128; j0 += 8) *(v8h*)&ar[j0] = pk;
        }
      }
      __syncthreads();
      if (mt < 7) {
        v8f acc[FOUTP / 16] = {};
#pragma unroll
        for (int kk = 0; kk < 4; ++kk) {
          v16h a = load_frag_rm(&att[wave][kk * 32], 128, lane);
#pragma unroll
          for (int nt = 0; nt < FOUTP / 16; ++nt) {
            v16h b = load_frag_rm(&hT[(nt * 16) * 128 + kk * 32], 128, lane);
            acc[nt] = wmma16(a, b, acc[nt]);
          }
        }
        int crow = (lane >> 4) * 8, ccol = lane & 15;
#pragma unroll
        for (int nt = 0; nt < FOUTP / 16; ++nt)
#pragma unroll
          for (int r = 0; r < 8; ++r) {
            int nd = mt * 16 + crow + r, c = nt * 16 + ccol;
            if (nd < kN && c < FOUT) {
              float v = acc[nt][r];
              if (CONCAT) {
                v += bl[hd * FOUT + c];
                v = v > 0.f ? v : (__expf(v) - 1.f);  // ELU
                out[((size_t)g * kN + nd) * OW + hd * FOUT + c] = v;
              } else {
                out[((size_t)g * kN + nd) * OW + c] = v + bl[c];
              }
            }
          }
      }
      __syncthreads();
    }
    __syncthreads();
  }
}

// ---------------------------------------------------------------------------
// 5) Node-mean pool + reparameterize. One 32-thread block per (b,t).
// ---------------------------------------------------------------------------
__global__ void k_pool(const float* __restrict__ e2o, const float* __restrict__ eps,
                       float* __restrict__ mu_out, float* __restrict__ lv_out,
                       float* __restrict__ z) {
  __shared__ float lg[32];
  int bt = blockIdx.x, c = threadIdx.x;
  float s = 0.f;
  for (int nd = 0; nd < kN; ++nd) s += e2o[((size_t)bt * kN + nd) * 32 + c];
  s *= (1.f / kN);
  lg[c] = s;
  if (c < kLAT) mu_out[bt * kLAT + c] = s;
  else          lv_out[bt * kLAT + (c - kLAT)] = s;
  __syncthreads();
  if (c < kLAT) z[bt * kLAT + c] = lg[c] + eps[bt * kLAT + c] * __expf(0.5f * lg[c + kLAT]);
}

__global__ void k_dcat(const float* __restrict__ emb, const float* __restrict__ z,
                       float* __restrict__ dc) {
  int i = blockIdx.x * 256 + threadIdx.x;
  if (i >= kRows * 80) return;
  int row = i / 80, c = i % 80;
  dc[i] = (c < 64) ? emb[(size_t)row * 64 + c] : z[(row / kN) * kLAT + (c - 64)];
}

__global__ void k_mask(const float* __restrict__ d3o, const float* __restrict__ m,
                       float* __restrict__ outp) {
  int i = blockIdx.x * 256 + threadIdx.x;
  if (i >= kRows * kACT) return;
  outp[i] = d3o[i] * m[i];
}

// ---------------------------------------------------------------------------
extern "C" void kernel_launch(void* const* d_in, const int* in_sizes, int n_in,
                              void* d_out, int out_size, void* d_ws, size_t ws_size,
                              hipStream_t stream) {
  (void)in_sizes; (void)n_in; (void)out_size; (void)ws_size;
  const float* o    = (const float*)d_in[0];
  const float* ap   = (const float*)d_in[1];
  const float* m    = (const float*)d_in[2];
  const float* eps  = (const float*)d_in[3];
  const int*   adj  = (const int*)  d_in[4];
  const float* Wo   = (const float*)d_in[5];
  const float* bo   = (const float*)d_in[6];
  const float* Wp   = (const float*)d_in[7];
  const float* bp   = (const float*)d_in[8];
  const float* Wm   = (const float*)d_in[9];
  const float* bm   = (const float*)d_in[10];
  const float* ide  = (const float*)d_in[11];
  const float* lWih = (const float*)d_in[12];
  const float* lWhh = (const float*)d_in[13];
  const float* lbih = (const float*)d_in[14];
  const float* lbhh = (const float*)d_in[15];
  const float* e1w = (const float*)d_in[16]; const float* e1s = (const float*)d_in[17];
  const float* e1d = (const float*)d_in[18]; const float* e1b = (const float*)d_in[19];
  const float* e2w = (const float*)d_in[20]; const float* e2s = (const float*)d_in[21];
  const float* e2d = (const float*)d_in[22]; const float* e2b = (const float*)d_in[23];
  const float* d1w = (const float*)d_in[24]; const float* d1s = (const float*)d_in[25];
  const float* d1d = (const float*)d_in[26]; const float* d1b = (const float*)d_in[27];
  const float* d2w = (const float*)d_in[28]; const float* d2s = (const float*)d_in[29];
  const float* d2d = (const float*)d_in[30]; const float* d2b = (const float*)d_in[31];
  const float* d3w = (const float*)d_in[32]; const float* d3s = (const float*)d_in[33];
  const float* d3d = (const float*)d_in[34]; const float* d3b = (const float*)d_in[35];

  char* ws = (char*)d_ws;
  float*     emb  = (float*)(ws + 0);              // 20.48 MB
  float*     ys   = (float*)(ws + 20480000);       // 20.48 MB
  float*     e1o  = (float*)(ws + 40960000);       // 20.48 MB
  float*     e2o  = (float*)(ws + 61440000);       // 10.24 MB
  float*     z    = (float*)(ws + 71680000);       // 51 KB
  _Float16*  xcat = (_Float16*)(ws + 71731200);    // 10.24 MB
  unsigned*  abits = (unsigned*)(ws + 81971200);   // 1792 B
  char*      gb   = ws + 81975296;                 // G region: 81.92 MB (recycled)
  float* G   = (float*)gb;
  float* dc  = (float*)gb;                         // after LSTM, G is dead
  float* d1o = (float*)(gb + 25600000);
  float* d2o = (float*)(gb + 46080000);
  float* d3o = (float*)(gb + 66560000);

  float* out    = (float*)d_out;                   // a_recon [16,50,100,8]
  float* mu_out = out + kRows * kACT;              // +640000
  float* lv_out = mu_out + kBS * kT * kLAT;        // +652800

  int nGraphs = kBS * kT;                          // 800
  k_xcat   <<<(kRows * 8 + 255) / 256, 256, 0, stream>>>(o, ap, m, xcat);
  k_adjbits<<<2, 256, 0, stream>>>(adj, abits);
  k_emb    <<<160, 256, 0, stream>>>(xcat, Wo, Wp, Wm, bo, bp, bm, ide, emb);
  k_gates  <<<320, 256, 0, stream>>>(emb, lWih, lbih, lbhh, G);
  k_lstm   <<<(kBS * kN) / 16, 128, 0, stream>>>(G, lWhh, ys);
  k_gat<64, 64, 4, 16, 16, true ><<<nGraphs, 128, 0, stream>>>(ys,  e1w, e1s, e1d, e1b, abits, e1o);
  k_gat<64, 64, 1, 32, 32, false><<<nGraphs, 128, 0, stream>>>(e1o, e2w, e2s, e2d, e2b, abits, e2o);
  k_pool   <<<nGraphs, 32, 0, stream>>>(e2o, eps, mu_out, lv_out, z);
  k_dcat   <<<(kRows * 80 + 255) / 256, 256, 0, stream>>>(emb, z, dc);
  k_gat<80, 96, 4, 16, 16, true ><<<nGraphs, 128, 0, stream>>>(dc,  d1w, d1s, d1d, d1b, abits, d1o);
  k_gat<64, 64, 4, 16, 16, true ><<<nGraphs, 128, 0, stream>>>(d1o, d2w, d2s, d2d, d2b, abits, d2o);
  k_gat<64, 64, 1,  8, 16, false><<<nGraphs, 128, 0, stream>>>(d2o, d3w, d3s, d3d, d3b, abits, d3o);
  k_mask   <<<(kRows * kACT + 255) / 256, 256, 0, stream>>>(d3o, m, out);
}
